// Decoder_2207613190319
// MI455X (gfx1250) — matrix operand order repaired and checked by reading, not compiled, not run on hardware
//
#include <hip/hip_runtime.h>
#include <hip/hip_bf16.h>
#include <math.h>

// ---- problem constants (match reference) ----
#define BB      8
#define LSEQ    512      // LD == LE == 512
#define DMODEL  512
#define NH      8
#define DKH     64
#define FFD     2048
#define NLAYER  6
#define NT      (BB * LSEQ)          // 4096 token rows
#define SCALE_QK 0.125f              // 1/sqrt(64)

typedef __attribute__((ext_vector_type(16))) __bf16 bf16x16;
typedef __attribute__((ext_vector_type(8)))  __bf16 bf16x8;
typedef __attribute__((ext_vector_type(4)))  __bf16 bf16x4;
typedef __attribute__((ext_vector_type(2)))  __bf16 bf16x2;
typedef __attribute__((ext_vector_type(8)))  float  f32x8;

__device__ __forceinline__ __bf16 f2bf(float f) {
  union { float f; unsigned u; } v; v.f = f;
  unsigned r = v.u + 0x7FFFu + ((v.u >> 16) & 1u);   // round-to-nearest-even
  unsigned short h = (unsigned short)(r >> 16);
  return __builtin_bit_cast(__bf16, h);
}

// packed f32x2 -> bf16x2 (v_cvt_pk_bf16_f32 if the target has it)
__device__ __forceinline__ bf16x2 cvt2(float a, float b) {
#if __has_builtin(__builtin_amdgcn_cvt_pk_bf16_f32)
  return __builtin_amdgcn_cvt_pk_bf16_f32(a, b);
#else
  bf16x2 r; r[0] = f2bf(a); r[1] = f2bf(b); return r;
#endif
}

// ---- CDNA5 async global->LDS path (ASYNCcnt), with safe fallback ----
#if defined(__HIP_DEVICE_COMPILE__) && __has_builtin(__builtin_amdgcn_global_load_async_to_lds_b128)
#define USE_ASYNC 1
#else
#define USE_ASYNC 0
#endif

typedef __attribute__((__vector_size__(16))) int i4v;
typedef __attribute__((address_space(1))) i4v g_i4v;   // "__device__" AS in HIP
typedef __attribute__((address_space(3))) i4v s_i4v;   // "__shared__" AS in HIP

__device__ __forceinline__ void cp16(void* l, const void* g) {
#if USE_ASYNC
  __builtin_amdgcn_global_load_async_to_lds_b128((g_i4v*)(void*)g, (s_i4v*)l, 0, 0);
#else
  *(float4*)l = *(const float4*)g;
#endif
}

#if USE_ASYNC
#if __has_builtin(__builtin_amdgcn_s_wait_asynccnt)
#define AWAIT(n) __builtin_amdgcn_s_wait_asynccnt(n)
#else
#define AWAIT(n) asm volatile("s_wait_asynccnt %0" ::"i"(n) : "memory")
#endif
#else
#define AWAIT(n) ((void)0)
#endif

__device__ __forceinline__ bf16x16 cvt16(const float* p0, const float* p1) {
  bf16x16 r;
#pragma unroll
  for (int i = 0; i < 4; ++i) {
    bf16x2 lo = cvt2(p0[2 * i], p0[2 * i + 1]);
    bf16x2 hi = cvt2(p1[2 * i], p1[2 * i + 1]);
    r[2 * i] = lo[0]; r[2 * i + 1] = lo[1];
    r[2 * i + 8] = hi[0]; r[2 * i + 9] = hi[1];
  }
  return r;
}

// ---------------------------------------------------------------------------
// WMMA GEMM: C[M,N] = A[M,K](f32) * B + (bias), optional ReLU.
// BMODE 0: B(k,n) = Bm[n*ldb + k]   (X @ W^T / Q @ K^T)
// BMODE 1: B(k,n) = Bm[k*ldb + n]   (w @ V)
// BT 1: B is pre-converted bf16 (weights) -> async byte-copy into LDS.
// BT 0: B is f32 (activations)           -> async raw copy, convert at frag.
// Double-buffered LDS: async DMA of chunk k+1 overlaps WMMAs of chunk k,
// synchronized with s_wait_asynccnt (per-wave in-order async completion).
// Block: 256 threads = 8 waves; tile 128(M) x 64(N); K-chunk 32.
// Each wave owns 16 rows x 64 cols = 4 x v_wmma_f32_16x16x32_bf16 per chunk.
// ---------------------------------------------------------------------------
template <int BMODE, int BT, bool BIAS, bool RELU>
__global__ __launch_bounds__(256) void gemm_wmma(
    const float* __restrict__ A, const void* __restrict__ Bmv,
    const float* __restrict__ bias, float* __restrict__ C,
    int M, int N, int K, int lda, int ldb, int ldc, int Hsplit,
    long long sAb, long long sAh, long long sBb, long long sBh,
    long long sCb, long long sCh)
{
  __shared__ __align__(16) float As[2][128 * 32];   // [m][k] raw f32
  __shared__ __align__(16) float Bs[2][64 * 32];    // BT0: f32 tile; BT1: bf16 alias

  const int tid  = threadIdx.x;
  const int lane = tid & 31;
  const int wave = tid >> 5;                  // 0..7 -> 16-row slab
  const int mBase = blockIdx.y * 128;
  const int nBase = blockIdx.x * 64;

  const int z    = blockIdx.z;
  const int bIdx = z / Hsplit;
  const int hIdx = z % Hsplit;
  A += (long long)bIdx * sAb + (long long)hIdx * sAh;
  C += (long long)bIdx * sCb + (long long)hIdx * sCh;
  const float*  Bf = (const float*)Bmv + (BT == 0 ? ((long long)bIdx * sBb + (long long)hIdx * sBh) : 0);
  const __bf16* Bw = (const __bf16*)Bmv + (BT == 1 ? ((long long)bIdx * sBb + (long long)hIdx * sBh) : 0);

  f32x8 acc[4] = {{}, {}, {}, {}};

  const int arow = tid >> 1;                  // 0..127
  const int acb  = (tid & 1) << 4;            // 0 / 16 floats

  // async instructions issued per wave per staged chunk
  constexpr short NB = (BMODE == 0 && BT == 1) ? 5 : 6;

  auto stage = [&](int it, int buf) {
    const int k0 = it << 5;
    {
      const float* src = &A[(long long)(mBase + arow) * lda + k0 + acb];
      float* dst = &As[buf][arow * 32 + acb];
      if (k0 + 32 < K) __builtin_prefetch(src + 32, 0, 3);   // global_prefetch_b8
      cp16(dst,      src);
      cp16(dst + 4,  src + 4);
      cp16(dst + 8,  src + 8);
      cp16(dst + 12, src + 12);
    }
    if (BMODE == 0) {
      const int n = tid >> 2;                 // 0..63
      const int c = (tid & 3) << 3;           // 0,8,16,24 elements
      if (BT == 1) {
        cp16((__bf16*)&Bs[buf][0] + n * 32 + c,
             &Bw[(long long)(nBase + n) * ldb + k0 + c]);
      } else {
        const float* src = &Bf[(long long)(nBase + n) * ldb + k0 + c];
        cp16(&Bs[buf][n * 32 + c],     src);
        cp16(&Bs[buf][n * 32 + c + 4], src + 4);
      }
    } else {
      const int kk = tid >> 3;                // 0..31
      const int c  = (tid & 7) << 3;          // 0..56 floats
      const float* src = &Bf[(long long)(k0 + kk) * ldb + nBase + c];
      cp16(&Bs[buf][kk * 64 + c],     src);
      cp16(&Bs[buf][kk * 64 + c + 4], src + 4);
    }
  };

  const int nIter = K >> 5;
  stage(0, 0);

  for (int it = 0; it < nIter; ++it) {
    const int cur = it & 1;
    // issue next chunk's DMA before consuming current (double buffer)
    if (it + 1 < nIter) { stage(it + 1, cur ^ 1); AWAIT(NB); }
    else                { AWAIT(0); }
    __syncthreads();

    // ---- fragments per CDNA5 wave32 VGPR layouts ----
    const float*  Asl = &As[cur][0];
    const float*  Bfl = &Bs[cur][0];
    const __bf16* Bbl = (const __bf16*)Bfl;

    const int kh   = lane >> 4;               // K half-group
    const int mrow = (wave << 4) + (lane & 15);
    const bf16x16 afrag = cvt16(&Asl[mrow * 32 + (kh << 4)],
                                &Asl[mrow * 32 + (kh << 4) + 8]);
#pragma unroll
    for (int t = 0; t < 4; ++t) {
      const int ncol = (t << 4) + (lane & 15);
      bf16x16 bfrag;
      if (BMODE == 0 && BT == 1) {
        bfrag = *(const bf16x16*)&Bbl[ncol * 32 + (kh << 4)];
      } else if (BMODE == 0) {
        bfrag = cvt16(&Bfl[ncol * 32 + (kh << 4)], &Bfl[ncol * 32 + (kh << 4) + 8]);
      } else {
        const int kb = kh << 4;
#pragma unroll
        for (int j = 0; j < 8; ++j) {
          bf16x2 p = cvt2(Bfl[(kb + 2 * j) * 64 + ncol], Bfl[(kb + 2 * j + 1) * 64 + ncol]);
          bfrag[2 * j] = p[0]; bfrag[2 * j + 1] = p[1];
        }
      }
      acc[t] = __builtin_amdgcn_wmma_f32_16x16x32_bf16(false, afrag, false, bfrag,
                                                       (short)0, acc[t], false, false);
    }
    __syncthreads();   // all waves done reading `cur` before it is re-staged
  }

  // ---- epilogue: D layout -> lane (N = lane&15, M = r + 8*(lane>>4)) ----
  const int kh = lane >> 4;
  const int cn = lane & 15;
#pragma unroll
  for (int t = 0; t < 4; ++t) {
    const int col = nBase + (t << 4) + cn;
    float bv = BIAS ? bias[col] : 0.f;
#pragma unroll
    for (int r = 0; r < 8; ++r) {
      const int row = mBase + (wave << 4) + r + (kh << 3);
      float v = acc[t][r] + bv;
      if (RELU) v = fmaxf(v, 0.f);
      C[(long long)row * ldc + col] = v;
    }
  }
}

// ---------------------------------------------------------------------------
// In-place masked softmax over rows of length 512; one wave per row.
// ---------------------------------------------------------------------------
__global__ __launch_bounds__(256) void softmax_mask_kernel(
    float* __restrict__ S, const int* __restrict__ toks,
    int Lq, int Lk, int causal, float scale)
{
  const int lane = threadIdx.x & 31;
  const int wave = threadIdx.x >> 5;
  const long long row = (long long)blockIdx.x * 8 + wave;
  const int q  = (int)(row % Lq);
  const int bh = (int)(row / Lq);
  const int b  = bh / NH;
  float* Sr = S + row * (long long)Lk;
  const int* tb = toks + (long long)b * Lk;

  float v[16];   // Lk = 512 = 32 lanes * 16
#pragma unroll
  for (int i = 0; i < 16; ++i) {
    const int k = lane + (i << 5);
    const float s = Sr[k];
    const bool m = (tb[k] == 0) || (causal && (k > q));
    v[i] = m ? -1e9f : s * scale;
  }
  float mx = v[0];
#pragma unroll
  for (int i = 1; i < 16; ++i) mx = fmaxf(mx, v[i]);
  for (int off = 16; off > 0; off >>= 1) mx = fmaxf(mx, __shfl_xor(mx, off, 32));
  float sum = 0.f;
#pragma unroll
  for (int i = 0; i < 16; ++i) { v[i] = __expf(v[i] - mx); sum += v[i]; }
  for (int off = 16; off > 0; off >>= 1) sum += __shfl_xor(sum, off, 32);
  const float inv = 1.0f / sum;
#pragma unroll
  for (int i = 0; i < 16; ++i) Sr[lane + (i << 5)] = v[i] * inv;
}

// ---------------------------------------------------------------------------
// out = LayerNorm(xin + res) * g + b   (row length DMODEL=512, wave per row)
// ---------------------------------------------------------------------------
__global__ __launch_bounds__(256) void add_ln_kernel(
    const float* __restrict__ xin, const float* __restrict__ res,
    const float* __restrict__ g, const float* __restrict__ bt,
    float* __restrict__ outp)
{
  const int lane = threadIdx.x & 31;
  const int wave = threadIdx.x >> 5;
  const long long row = (long long)blockIdx.x * 8 + wave;
  const float* xr = xin + row * DMODEL;
  const float* rr = res + row * DMODEL;
  float* orow = outp + row * DMODEL;

  float v[16], sum = 0.f, sq = 0.f;
#pragma unroll
  for (int i = 0; i < 16; ++i) {
    const int d = lane + (i << 5);
    const float t = xr[d] + rr[d];
    v[i] = t; sum += t; sq += t * t;
  }
  for (int off = 16; off > 0; off >>= 1) {
    sum += __shfl_xor(sum, off, 32);
    sq  += __shfl_xor(sq,  off, 32);
  }
  const float mean = sum * (1.0f / DMODEL);
  const float var  = sq * (1.0f / DMODEL) - mean * mean;
  const float rstd = rsqrtf(var + 1e-5f);
#pragma unroll
  for (int i = 0; i < 16; ++i) {
    const int d = lane + (i << 5);
    orow[d] = (v[i] - mean) * rstd * g[d] + bt[d];
  }
}

// ---------------------------------------------------------------------------
// X[b,l,:] = emb[dec[b,l],:] + sinusoidal PE(pos = l+1)
// ---------------------------------------------------------------------------
__global__ __launch_bounds__(256) void embed_kernel(
    const int* __restrict__ dec, const float* __restrict__ emb,
    float* __restrict__ X)
{
  const long long i = (long long)blockIdx.x * 256 + threadIdx.x;
  const int d = (int)(i % DMODEL);
  const long long bl = i / DMODEL;
  const int l = (int)(bl % LSEQ);
  const int tok = dec[bl];
  const int j2 = d & ~1;
  const float angle = (float)(l + 1) *
                      __powf(10000.0f, -(float)j2 / (float)DMODEL);
  const float pe = (d & 1) ? __cosf(angle) : __sinf(angle);
  X[i] = emb[(long long)tok * DMODEL + d] + pe;
}

// ---------------------------------------------------------------------------
// One-shot f32 -> bf16 weight image (4 elements / thread)
// ---------------------------------------------------------------------------
__global__ __launch_bounds__(256) void cvt_bf16_kernel(
    const float* __restrict__ src, __bf16* __restrict__ dst)
{
  const long long i = ((long long)blockIdx.x * 256 + threadIdx.x) * 4;
  float4 f = *(const float4*)(src + i);
  bf16x2 lo = cvt2(f.x, f.y);
  bf16x2 hi = cvt2(f.z, f.w);
  bf16x4 t;
  t[0] = lo[0]; t[1] = lo[1]; t[2] = hi[0]; t[3] = hi[1];
  *(bf16x4*)(dst + i) = t;
}

// ---------------------------------------------------------------------------
// Host-side orchestration
// ---------------------------------------------------------------------------
static void run_attention(hipStream_t stream,
                          float* X, const float* kvsrc, const int* toks, int causal,
                          const __bf16* wq, const float* bq,
                          const __bf16* wk, const float* bk,
                          const __bf16* wv, const float* bv,
                          const __bf16* wo, const float* bo,
                          const float* g, const float* bt,
                          float* QB, float* KB, float* VB,
                          float* CTX, float* TMP, float* Wout)
{
  const dim3 blk(256);
  const dim3 gProj(DMODEL / 64, NT / 128, 1);
  // Q/K/V projections (X @ W^T + b), bf16 weight image
  gemm_wmma<0, 1, true, false><<<gProj, blk, 0, stream>>>(
      X, wq, bq, QB, NT, DMODEL, DMODEL, DMODEL, DMODEL, DMODEL, 1, 0, 0, 0, 0, 0, 0);
  gemm_wmma<0, 1, true, false><<<gProj, blk, 0, stream>>>(
      kvsrc, wk, bk, KB, NT, DMODEL, DMODEL, DMODEL, DMODEL, DMODEL, 1, 0, 0, 0, 0, 0, 0);
  gemm_wmma<0, 1, true, false><<<gProj, blk, 0, stream>>>(
      kvsrc, wv, bv, VB, NT, DMODEL, DMODEL, DMODEL, DMODEL, DMODEL, 1, 0, 0, 0, 0, 0, 0);

  // scores: per (b,h)  S[q,k] = Q . K   (raw; scale+mask in softmax)
  gemm_wmma<0, 0, false, false><<<dim3(LSEQ / 64, LSEQ / 128, BB * NH), blk, 0, stream>>>(
      QB, KB, nullptr, Wout, LSEQ, LSEQ, DKH,
      DMODEL, DMODEL, LSEQ, NH,
      (long long)LSEQ * DMODEL, DKH,
      (long long)LSEQ * DMODEL, DKH,
      (long long)NH * LSEQ * LSEQ, (long long)LSEQ * LSEQ);

  softmax_mask_kernel<<<(BB * NH * LSEQ) / 8, blk, 0, stream>>>(
      Wout, toks, LSEQ, LSEQ, causal, SCALE_QK);

  // ctx: per (b,h)  CTX[q,d] = sum_k W[q,k] * V[k,d]
  gemm_wmma<1, 0, false, false><<<dim3(1, LSEQ / 128, BB * NH), blk, 0, stream>>>(
      Wout, VB, nullptr, CTX, LSEQ, DKH, LSEQ,
      LSEQ, DMODEL, DMODEL, NH,
      (long long)NH * LSEQ * LSEQ, (long long)LSEQ * LSEQ,
      (long long)LSEQ * DMODEL, DKH,
      (long long)LSEQ * DMODEL, DKH);

  // output projection + residual LayerNorm
  gemm_wmma<0, 1, true, false><<<gProj, blk, 0, stream>>>(
      CTX, wo, bo, TMP, NT, DMODEL, DMODEL, DMODEL, DMODEL, DMODEL, 1, 0, 0, 0, 0, 0, 0);
  add_ln_kernel<<<NT / 8, blk, 0, stream>>>(TMP, X, g, bt, X);
}

extern "C" void kernel_launch(void* const* d_in, const int* in_sizes, int n_in,
                              void* d_out, int out_size, void* d_ws, size_t ws_size,
                              hipStream_t stream)
{
  (void)in_sizes; (void)n_in; (void)out_size; (void)ws_size;

  const int*   dec  = (const int*)d_in[0];
  const int*   enc  = (const int*)d_in[1];
  const float* encO = (const float*)d_in[2];
  const float* embT = (const float*)d_in[3];
  const float* sa_wq = (const float*)d_in[4];
  const float* sa_wk = (const float*)d_in[5];
  const float* sa_wv = (const float*)d_in[6];
  const float* sa_wo = (const float*)d_in[7];
  const float* sa_bq = (const float*)d_in[8];
  const float* sa_bk = (const float*)d_in[9];
  const float* sa_bv = (const float*)d_in[10];
  const float* sa_bo = (const float*)d_in[11];
  const float* sa_g  = (const float*)d_in[12];
  const float* sa_b  = (const float*)d_in[13];
  const float* ca_wq = (const float*)d_in[14];
  const float* ca_wk = (const float*)d_in[15];
  const float* ca_wv = (const float*)d_in[16];
  const float* ca_wo = (const float*)d_in[17];
  const float* ca_bq = (const float*)d_in[18];
  const float* ca_bk = (const float*)d_in[19];
  const float* ca_bv = (const float*)d_in[20];
  const float* ca_bo = (const float*)d_in[21];
  const float* ca_g  = (const float*)d_in[22];
  const float* ca_b  = (const float*)d_in[23];
  const float* f_w1  = (const float*)d_in[24];
  const float* f_b1  = (const float*)d_in[25];
  const float* f_w2  = (const float*)d_in[26];
  const float* f_b2  = (const float*)d_in[27];
  const float* f_g   = (const float*)d_in[28];
  const float* f_b   = (const float*)d_in[29];

  const long long XSZ  = (long long)NT * DMODEL;            // 2,097,152
  const long long AWSZ = (long long)BB * NH * LSEQ * LSEQ;  // per-layer attn weights

  float* out       = (float*)d_out;
  float* sa_ws_out = out + XSZ;
  float* ca_ws_out = sa_ws_out + (long long)NLAYER * AWSZ;

  // ---- workspace: 6 f32 activation buffers + bf16 weight image ----
  float* ws   = (float*)d_ws;
  float* X    = ws;
  float* QB   = X   + XSZ;
  float* KB   = QB  + XSZ;
  float* VB   = KB  + XSZ;
  float* CTX  = VB  + XSZ;
  float* TMP  = CTX + XSZ;
  float* FMID = QB;                // FFN mid aliases idle Q/K/V/CTX (4*XSZ floats)

  const long long WATT = (long long)NLAYER * DMODEL * DMODEL;   // per attn tensor
  const long long WFFN = (long long)NLAYER * FFD * DMODEL;      // per ffn tensor
  __bf16* wb = (__bf16*)(ws + 6 * XSZ);
  __bf16* sa_wq_b = wb + 0 * WATT;
  __bf16* sa_wk_b = wb + 1 * WATT;
  __bf16* sa_wv_b = wb + 2 * WATT;
  __bf16* sa_wo_b = wb + 3 * WATT;
  __bf16* ca_wq_b = wb + 4 * WATT;
  __bf16* ca_wk_b = wb + 5 * WATT;
  __bf16* ca_wv_b = wb + 6 * WATT;
  __bf16* ca_wo_b = wb + 7 * WATT;
  __bf16* f_w1_b  = wb + 8 * WATT;
  __bf16* f_w2_b  = f_w1_b + WFFN;

  const dim3 blk(256);

  // one-shot weight conversion to bf16
  cvt_bf16_kernel<<<(unsigned)(WATT / 1024), blk, 0, stream>>>(sa_wq, sa_wq_b);
  cvt_bf16_kernel<<<(unsigned)(WATT / 1024), blk, 0, stream>>>(sa_wk, sa_wk_b);
  cvt_bf16_kernel<<<(unsigned)(WATT / 1024), blk, 0, stream>>>(sa_wv, sa_wv_b);
  cvt_bf16_kernel<<<(unsigned)(WATT / 1024), blk, 0, stream>>>(sa_wo, sa_wo_b);
  cvt_bf16_kernel<<<(unsigned)(WATT / 1024), blk, 0, stream>>>(ca_wq, ca_wq_b);
  cvt_bf16_kernel<<<(unsigned)(WATT / 1024), blk, 0, stream>>>(ca_wk, ca_wk_b);
  cvt_bf16_kernel<<<(unsigned)(WATT / 1024), blk, 0, stream>>>(ca_wv, ca_wv_b);
  cvt_bf16_kernel<<<(unsigned)(WATT / 1024), blk, 0, stream>>>(ca_wo, ca_wo_b);
  cvt_bf16_kernel<<<(unsigned)(WFFN / 1024), blk, 0, stream>>>(f_w1, f_w1_b);
  cvt_bf16_kernel<<<(unsigned)(WFFN / 1024), blk, 0, stream>>>(f_w2, f_w2_b);

  // embedding + positional encoding
  embed_kernel<<<(unsigned)(XSZ / 256), blk, 0, stream>>>(dec, embT, X);

  for (int l = 0; l < NLAYER; ++l) {
    const long long wOff = (long long)l * DMODEL * DMODEL;
    const long long bOff = (long long)l * DMODEL;

    // ---- self-attention (causal + dec pad mask) ----
    run_attention(stream, X, X, dec, /*causal=*/1,
                  sa_wq_b + wOff, sa_bq + bOff, sa_wk_b + wOff, sa_bk + bOff,
                  sa_wv_b + wOff, sa_bv + bOff, sa_wo_b + wOff, sa_bo + bOff,
                  sa_g + bOff, sa_b + bOff,
                  QB, KB, VB, CTX, TMP, sa_ws_out + (long long)l * AWSZ);

    // ---- cross-attention (enc pad mask) ----
    run_attention(stream, X, encO, enc, /*causal=*/0,
                  ca_wq_b + wOff, ca_bq + bOff, ca_wk_b + wOff, ca_bk + bOff,
                  ca_wv_b + wOff, ca_bv + bOff, ca_wo_b + wOff, ca_bo + bOff,
                  ca_g + bOff, ca_b + bOff,
                  QB, KB, VB, CTX, TMP, ca_ws_out + (long long)l * AWSZ);

    // ---- FFN ----
    const __bf16* w1 = f_w1_b + (long long)l * FFD * DMODEL;
    const float*  b1 = f_b1 + (long long)l * FFD;
    const __bf16* w2 = f_w2_b + (long long)l * DMODEL * FFD;
    const float*  b2 = f_b2 + (long long)l * DMODEL;
    gemm_wmma<0, 1, true, true><<<dim3(FFD / 64, NT / 128, 1), blk, 0, stream>>>(
        X, w1, b1, FMID, NT, FFD, DMODEL, DMODEL, DMODEL, FFD, 1, 0, 0, 0, 0, 0, 0);
    gemm_wmma<0, 1, true, false><<<dim3(DMODEL / 64, NT / 128, 1), blk, 0, stream>>>(
        FMID, w2, b2, TMP, NT, DMODEL, FFD, FFD, FFD, DMODEL, 1, 0, 0, 0, 0, 0, 0);
    add_ln_kernel<<<NT / 8, blk, 0, stream>>>(TMP, X, f_g + bOff, f_b + bOff, X);
  }

  // final hidden states -> d_out[0 : NT*DMODEL]
  (void)hipMemcpyAsync(d_out, X, (size_t)XSZ * sizeof(float),
                       hipMemcpyDeviceToDevice, stream);
}
